// NodeLevelEncoder_53961969107354
// MI455X (gfx1250) — matrix operand order repaired and checked by reading, not compiled, not run on hardware
//
#include <hip/hip_runtime.h>
#include <hip/hip_bf16.h>

// ---------------- problem constants (match reference) ----------------
#define NN 100000
#define EE 800000
#define GG 64
#define HL 256

typedef __attribute__((ext_vector_type(2)))  _Float16 v2h;
typedef __attribute__((ext_vector_type(4)))  _Float16 v4h;
typedef __attribute__((ext_vector_type(8)))  _Float16 v8h;
typedef __attribute__((ext_vector_type(16))) _Float16 v16h;
typedef __attribute__((ext_vector_type(8)))  float    v8f;

#if defined(__has_builtin)
#if __has_builtin(__builtin_amdgcn_tensor_load_to_lds) && __has_builtin(__builtin_amdgcn_s_wait_tensorcnt)
#define HAVE_TDM 1
#endif
#endif
#ifndef HAVE_TDM
#define HAVE_TDM 0
#endif

#if HAVE_TDM
typedef __attribute__((ext_vector_type(4))) unsigned int v4u;
typedef __attribute__((ext_vector_type(8))) int          v8i_t;
typedef __attribute__((ext_vector_type(4))) int          v4i_t;

// ---------------------------------------------------------------------
// TDM 2D tile load: `rows` x 32 f16 tile, row stride `stride_halves`,
// into LDS at byte offset lds_off with pitch 34 halves (pad 1 DWORD per
// 16 DWORDs via D#.pad_*). D# layout per CDNA5 ISA section 8.
// This toolchain's builtin is the 6-arg form:
//   (uint32x4 g0, int32x8 g1, int32x4 g2, int32x4 g3, int32x8 g4, i32 cpol)
// ---------------------------------------------------------------------
__device__ __forceinline__ void tdm_load_tile(unsigned lds_off, const _Float16* g,
                                              unsigned rows, unsigned stride_halves)
{
    unsigned long long ga = (unsigned long long)(size_t)g;
    v4u g0;
    g0[0] = 1u;                                    // count=1 valid user descriptor
    g0[1] = lds_off;                               // lds_addr (bytes)
    g0[2] = (unsigned)ga;                          // global_addr[31:0]
    g0[3] = (unsigned)((ga >> 32) & 0x01FFFFFFu)   // global_addr[56:32]
            | 0x80000000u;                         // type = 2 ("image")
    v8i_t g1;
    g1[0] = (1 << 16)                              // data_size = 2 bytes
          | (1 << 20)                              // pad_enable
          | (3 << 22)                              // pad_interval: every 16 DWORDs (one 32-half row)
          | (0 << 25);                             // pad_amount: 1 DWORD -> pitch 34 halves
    g1[1] = 0;                                     // tensor_dim0[15:0] (dim0 = 1<<20)
    g1[2] = 16;                                    // tensor_dim0[31:16]=16 ; tensor_dim1[15:0]=0
    g1[3] = 16 | (32 << 16);                       // tensor_dim1[31:16]=16 ; tile_dim0 = 32
    g1[4] = (int)rows;                             // tile_dim1 = rows ; tile_dim2 = 0
    g1[5] = (int)stride_halves;                    // tensor_dim0_stride[31:0]
    g1[6] = 0;
    g1[7] = 0;
    v4i_t z4 = {0, 0, 0, 0};
    v8i_t z8 = {0, 0, 0, 0, 0, 0, 0, 0};
    __builtin_amdgcn_tensor_load_to_lds(g0, g1, z4, z4, z8, 0);
}
#endif

// ---- fragment build (ISA 7.12.2 packing) + 2 WMMAs, shared by both paths ----
__device__ __forceinline__ void frag_mma(const _Float16* sA, const _Float16* sB,
                                         int m, int sel, int wave, v8f& acc0, v8f& acc1)
{
    v16h a0, a1, b;
    #pragma unroll
    for (int v = 0; v < 8; ++v) {
        // A 16x32: lanes hold M; K = (v<4 ? 0:16) + sel*8 + 2(v&3)+p
        int ka = ((v & 4) ? 16 : 0) + sel * 8 + (v & 3) * 2;
        v2h pa0 = *(const v2h*)&sA[m * 34 + ka];
        v2h pa1 = *(const v2h*)&sA[(16 + m) * 34 + ka];
        a0[2 * v] = pa0[0]; a0[2 * v + 1] = pa0[1];
        a1[2 * v] = pa1[0]; a1[2 * v + 1] = pa1[1];
        // B 32x16: lane = N; K = (v<4 ? 0:16) + sel*8 + 2(v&3)+p
        int kb = ((v & 4) ? 16 : 0) + sel * 8 + (v & 3) * 2;
        v2h pb = *(const v2h*)&sB[(wave * 16 + m) * 34 + kb];
        b[2 * v] = pb[0]; b[2 * v + 1] = pb[1];
    }
    acc0 = __builtin_amdgcn_wmma_f32_16x16x32_f16(false, a0, false, b, (short)0, acc0, false, false);
    acc1 = __builtin_amdgcn_wmma_f32_16x16x32_f16(false, a1, false, b, (short)0, acc1, false, false);
}

// ---------------------------------------------------------------------
// WMMA GEMM:  C[M x Nout](f32) = act( A[M x K](f16) * B[Nout x K](f16)ᵀ + bias )
// Block: 256 threads / 8 waves; tile 32 rows x 128 cols; each wave: two
// 16x16 row-tiles sharing one B fragment -> 2 WMMA per k-step.
// TDM path: A(32x32) and B(128x32) tiles DMA'd to LDS by wave 0 with
// tensor_load_to_lds, double-buffered so the copy of tile t+1 overlaps
// the WMMAs of tile t; sync via s_wait_tensorcnt + workgroup barrier.
// Requires M%32==0, K%32==0, Nout%128==0 (true for all calls here).
// ---------------------------------------------------------------------
__global__ __launch_bounds__(256)
void k_gemm_wmma(const _Float16* __restrict__ A, const _Float16* __restrict__ B,
                 const float* __restrict__ bias, float* __restrict__ C,
                 int M, int K, int Nout, int act)
{
    const int tid  = threadIdx.x;
    const int wave = tid >> 5;
    const int lane = tid & 31;
    const int row0 = blockIdx.x * 32;
    const int colB = blockIdx.y * 128;
    const int col  = colB + wave * 16;
    const int m    = lane & 15;
    const int sel  = lane >> 4;

    v8f acc0 = {}, acc1 = {};

#if HAVE_TDM
    __shared__ _Float16 sA[2][32 * 34];
    __shared__ _Float16 sB[2][128 * 34];
    const int nk = K >> 5;
    if (wave == 0) {
        tdm_load_tile((unsigned)(size_t)&sA[0][0], A + (size_t)row0 * K, 32, K);
        tdm_load_tile((unsigned)(size_t)&sB[0][0], B + (size_t)colB * K, 128, K);
        __builtin_amdgcn_s_wait_tensorcnt(0);
    }
    __syncthreads();
    for (int t = 0; t < nk; ++t) {
        const int cur = t & 1;
        const int nxt = cur ^ 1;
        if (wave == 0 && t + 1 < nk) {   // overlap next tile DMA with WMMAs
            tdm_load_tile((unsigned)(size_t)&sA[nxt][0],
                          A + (size_t)row0 * K + (t + 1) * 32, 32, K);
            tdm_load_tile((unsigned)(size_t)&sB[nxt][0],
                          B + (size_t)colB * K + (t + 1) * 32, 128, K);
        }
        frag_mma(sA[cur], sB[cur], m, sel, wave, acc0, acc1);
        __syncthreads();                                  // done reading cur
        if (wave == 0 && t + 1 < nk) __builtin_amdgcn_s_wait_tensorcnt(0);
        __syncthreads();                                  // nxt ready
    }
#else
    __shared__ _Float16 sA1[32 * 34];
    __shared__ _Float16 sB1[128 * 34];
    for (int k0 = 0; k0 < K; k0 += 32) {
        {   // stage A 32x32 halves: thread = 4 consecutive halves of one row
            int e = tid * 4, r = e >> 5, kk = e & 31;
            v4h va = *(const v4h*)&A[(size_t)(row0 + r) * K + k0 + kk];
            v2h p0 = {va[0], va[1]}, p1 = {va[2], va[3]};
            *(v2h*)&sA1[r * 34 + kk]     = p0;
            *(v2h*)&sA1[r * 34 + kk + 2] = p1;
        }
        {   // stage B 128x32 halves: thread = 16 consecutive halves of one row
            int n = tid >> 1, kk = (tid & 1) * 16;
            const _Float16* src = &B[(size_t)(colB + n) * K + k0 + kk];
            #pragma unroll
            for (int q = 0; q < 2; ++q) {
                v8h vb = *(const v8h*)(src + 8 * q);
                #pragma unroll
                for (int p = 0; p < 4; ++p) {
                    v2h pp = {vb[2 * p], vb[2 * p + 1]};
                    *(v2h*)&sB1[n * 34 + kk + 8 * q + 2 * p] = pp;
                }
            }
        }
        __syncthreads();
        frag_mma(sA1, sB1, m, sel, wave, acc0, acc1);
        __syncthreads();
    }
#endif

    // epilogue: C vgpr j -> row = 8*sel + j, col = lane & 15
    const float bv = bias ? bias[col + m] : 0.0f;
    #pragma unroll
    for (int j = 0; j < 8; ++j) {
        int rr = row0 + sel * 8 + j;
        float v0 = acc0[j] + bv;
        float v1 = acc1[j] + bv;
        if (act) { v0 = fmaxf(v0, 0.0f); v1 = fmaxf(v1, 0.0f); }
        C[(size_t)rr * Nout + col + m]        = v0;
        C[(size_t)(rr + 16) * Nout + col + m] = v1;
    }
}

// ---------------- conversion / elementwise / graph kernels ----------------
__global__ void k_fill(float* p, long long n, float v) {
    long long i = (long long)blockIdx.x * blockDim.x + threadIdx.x;
    if (i < n) p[i] = v;
}

__global__ void k_cvt(_Float16* __restrict__ o, const float* __restrict__ a, long long n) {
    long long i = (long long)blockIdx.x * blockDim.x + threadIdx.x;
    if (i < n) o[i] = (_Float16)a[i];
}

// o[n*K + k] = (f16) a[k*N + n]   (K x N row-major -> N x K row-major)
__global__ void k_transpose_cvt(_Float16* __restrict__ o, const float* __restrict__ a,
                                int K, int N) {
    int i = blockIdx.x * blockDim.x + threadIdx.x;
    if (i < K * N) {
        int n = i / K, k = i % K;
        o[i] = (_Float16)a[(size_t)k * N + n];
    }
}

__global__ void k_deg(const long long* __restrict__ ei, float* __restrict__ deg) {
    int e = blockIdx.x * blockDim.x + threadIdx.x;
    if (e < EE) atomicAdd(&deg[(int)ei[EE + e]], 1.0f);
}

__global__ void k_rsqrt(const float* __restrict__ deg, float* __restrict__ dinv) {
    int i = blockIdx.x * blockDim.x + threadIdx.x;
    if (i < NN) dinv[i] = rsqrtf(deg[i]);
}

__global__ void k_edge_norm(const long long* __restrict__ ei,
                            const float* __restrict__ dinv,
                            float* __restrict__ enorm) {
    int e = blockIdx.x * blockDim.x + threadIdx.x;
    if (e < EE) enorm[e] = dinv[(int)ei[e]] * dinv[(int)ei[EE + e]];
}

// seed accumulator with self-loop contribution + bias (aliases h buffer)
__global__ void k_seed(float* __restrict__ accum, const float* __restrict__ T,
                       const float* __restrict__ dinv, const float* __restrict__ bias,
                       int D) {
    int n = blockIdx.x;
    float s = dinv[n] * dinv[n];
    for (int f = threadIdx.x; f < D; f += blockDim.x)
        accum[(size_t)n * D + f] = T[(size_t)n * D + f] * s + bias[f];
}

// one block per edge: float4 gather of transformed src features, f32 atomics to dst
__global__ void k_aggregate(const long long* __restrict__ ei,
                            const float* __restrict__ enorm,
                            const float* __restrict__ T,
                            float* __restrict__ accum, int D) {
    int e   = blockIdx.x;
    int src = (int)ei[e];
    int dst = (int)ei[EE + e];
    float w = enorm[e];
    int c = threadIdx.x * 4;
    if (c < D) {
        const float4 v = *(const float4*)&T[(size_t)src * D + c];
        float* o = &accum[(size_t)dst * D + c];
        atomicAdd(o + 0, v.x * w);
        atomicAdd(o + 1, v.y * w);
        atomicAdd(o + 2, v.z * w);
        atomicAdd(o + 3, v.w * w);
    }
}

// relu in place (f32, feeds pooling) + f16 copy (feeds next WMMA GEMM)
__global__ void k_relu_cvt(float* __restrict__ p, _Float16* __restrict__ q, long long n) {
    long long i = (long long)blockIdx.x * blockDim.x + threadIdx.x;
    if (i < n) {
        float v = fmaxf(p[i], 0.0f);
        p[i] = v;
        q[i] = (_Float16)v;
    }
}

__global__ void k_count(const long long* __restrict__ batch, float* __restrict__ counts) {
    int n = blockIdx.x * blockDim.x + threadIdx.x;
    if (n < NN) atomicAdd(&counts[(int)batch[n]], 1.0f);
}

__global__ void k_pool_sum(const float* __restrict__ h, const long long* __restrict__ batch,
                           float* __restrict__ pooled) {
    int n = blockIdx.x;
    int g = (int)batch[n];
    int c = threadIdx.x * 4;
    const float4 v = *(const float4*)&h[(size_t)n * 512 + c];
    float* o = &pooled[(size_t)g * 512 + c];
    atomicAdd(o + 0, v.x);
    atomicAdd(o + 1, v.y);
    atomicAdd(o + 2, v.z);
    atomicAdd(o + 3, v.w);
}

__global__ void k_pool_div_cvt(_Float16* __restrict__ o, const float* __restrict__ pooled,
                               const float* __restrict__ counts) {
    int i = blockIdx.x * blockDim.x + threadIdx.x;
    if (i < GG * 512) o[i] = (_Float16)(pooled[i] / fmaxf(counts[i >> 9], 1.0f));
}

__global__ void k_vec_add(float* __restrict__ o, const float* __restrict__ a,
                          const float* __restrict__ b, int n) {
    int i = blockIdx.x * blockDim.x + threadIdx.x;
    if (i < n) o[i] = a[i] + b[i];
}

__device__ __forceinline__ float sigm(float x) { return 1.0f / (1.0f + __expf(-x)); }

// zero initial state: c = i*g, h = o*tanh(c)   (gates order i,f,g,o); f16 out
__global__ void k_lstm(const float* __restrict__ gates, _Float16* __restrict__ h) {
    int i = blockIdx.x * blockDim.x + threadIdx.x;
    if (i < GG * HL) {
        int b = i >> 8, j = i & 255;
        const float* g4 = &gates[(size_t)b * 4 * HL];
        float ig = sigm(g4[j]);
        float gg = tanhf(g4[2 * HL + j]);
        float og = sigm(g4[3 * HL + j]);
        h[i] = (_Float16)(og * tanhf(ig * gg));
    }
}

// ---------------------------------------------------------------------
extern "C" void kernel_launch(void* const* d_in, const int* in_sizes, int n_in,
                              void* d_out, int out_size, void* d_ws, size_t ws_size,
                              hipStream_t stream)
{
    const float*     x     = (const float*)d_in[0];
    const long long* ei    = (const long long*)d_in[1];
    const long long* batch = (const long long*)d_in[2];
    const float* W1 = (const float*)d_in[3],  *b1 = (const float*)d_in[4];
    const float* W2 = (const float*)d_in[5],  *b2 = (const float*)d_in[6];
    const float* W3 = (const float*)d_in[7],  *b3 = (const float*)d_in[8];
    const float* Wih0 = (const float*)d_in[9];  // [1024,512]
    const float* bih0 = (const float*)d_in[11], *bhh0 = (const float*)d_in[12];
    const float* Wih1 = (const float*)d_in[13]; // [1024,256]
    const float* bih1 = (const float*)d_in[15], *bhh1 = (const float*)d_in[16];
    const float* Wp = (const float*)d_in[17],  *bp = (const float*)d_in[18];

    // ---- f32 workspace ----
    float* ws = (float*)d_ws;
    size_t off = 0;
    float* hbuf   = ws + off; off += (size_t)NN * 512;  // layer out / agg accumulator
    float* T      = ws + off; off += (size_t)NN * 512;  // transformed features
    float* deg    = ws + off; off += NN;
    float* dinv   = ws + off; off += NN;
    float* enorm  = ws + off; off += EE;
    float* pooled = ws + off; off += (size_t)GG * 512;
    float* counts = ws + off; off += GG;
    float* gates  = ws + off; off += (size_t)GG * 1024;
    float* bsum   = ws + off; off += 1024;

    // ---- f16 workspace (starts 64B aligned) ----
    _Float16* f16ws = (_Float16*)(ws + off);
    size_t ho = 0;
    _Float16* x16      = f16ws + ho; ho += (size_t)NN * 32;
    _Float16* h16      = f16ws + ho; ho += (size_t)NN * 512;
    _Float16* w1t      = f16ws + ho; ho += 128 * 32;
    _Float16* w2t      = f16ws + ho; ho += 256 * 128;
    _Float16* w3t      = f16ws + ho; ho += 512 * 256;
    _Float16* wih0h    = f16ws + ho; ho += 1024 * 512;
    _Float16* wih1h    = f16ws + ho; ho += 1024 * 256;
    _Float16* wph      = f16ws + ho; ho += 512 * 256;
    _Float16* pooled16 = f16ws + ho; ho += (size_t)GG * 512;
    _Float16* h1h      = f16ws + ho; ho += (size_t)GG * HL;
    _Float16* h2h      = f16ws + ho; ho += (size_t)GG * HL;

    // ---- one-time precision conversions (weights + input features) ----
    k_cvt<<<(NN * 32 + 255) / 256, 256, 0, stream>>>(x16, x, (long long)NN * 32);
    k_transpose_cvt<<<(32 * 128 + 255) / 256, 256, 0, stream>>>(w1t, W1, 32, 128);
    k_transpose_cvt<<<(128 * 256 + 255) / 256, 256, 0, stream>>>(w2t, W2, 128, 256);
    k_transpose_cvt<<<(256 * 512 + 255) / 256, 256, 0, stream>>>(w3t, W3, 256, 512);
    k_cvt<<<(1024 * 512 + 255) / 256, 256, 0, stream>>>(wih0h, Wih0, 1024 * 512);
    k_cvt<<<(1024 * 256 + 255) / 256, 256, 0, stream>>>(wih1h, Wih1, 1024 * 256);
    k_cvt<<<(512 * 256 + 255) / 256, 256, 0, stream>>>(wph, Wp, 512 * 256);

    // ---- degrees / normalization ----
    k_fill<<<(NN + 255) / 256, 256, 0, stream>>>(deg, NN, 1.0f);           // self loop
    k_deg<<<(EE + 255) / 256, 256, 0, stream>>>(ei, deg);
    k_rsqrt<<<(NN + 255) / 256, 256, 0, stream>>>(deg, dinv);
    k_edge_norm<<<(EE + 255) / 256, 256, 0, stream>>>(ei, dinv, enorm);

    // ---- 3 GCN layers (M = 100000 = 3125 * 32) ----
    const _Float16* lin16 = x16;
    const _Float16* lWt[3] = {w1t, w2t, w3t};
    const float*    lb[3]  = {b1, b2, b3};
    int lK[3] = {32, 128, 256};
    int lD[3] = {128, 256, 512};
    for (int L = 0; L < 3; ++L) {
        int K = lK[L], D = lD[L];
        k_gemm_wmma<<<dim3(NN / 32, D / 128), 256, 0, stream>>>(
            lin16, lWt[L], nullptr, T, NN, K, D, 0);
        k_seed<<<NN, 128, 0, stream>>>(hbuf, T, dinv, lb[L], D);
        k_aggregate<<<EE, D / 4, 0, stream>>>(ei, enorm, T, hbuf, D);
        k_relu_cvt<<<(int)(((long long)NN * D + 255) / 256), 256, 0, stream>>>(
            hbuf, h16, (long long)NN * D);
        lin16 = h16;
    }

    // ---- global mean pool ----
    k_fill<<<1, 64, 0, stream>>>(counts, GG, 0.0f);
    k_fill<<<(GG * 512 + 255) / 256, 256, 0, stream>>>(pooled, GG * 512, 0.0f);
    k_count<<<(NN + 255) / 256, 256, 0, stream>>>(batch, counts);
    k_pool_sum<<<NN, 128, 0, stream>>>(hbuf, batch, pooled);
    k_pool_div_cvt<<<(GG * 512 + 255) / 256, 256, 0, stream>>>(pooled16, pooled, counts);

    // ---- LSTM layer 0 (h0=c0=0 => gates = pooled @ Wih0^T + bih0 + bhh0) ----
    k_vec_add<<<4, 256, 0, stream>>>(bsum, bih0, bhh0, 1024);
    k_gemm_wmma<<<dim3(GG / 32, 1024 / 128), 256, 0, stream>>>(
        pooled16, wih0h, bsum, gates, GG, 512, 1024, 0);
    k_lstm<<<(GG * HL + 255) / 256, 256, 0, stream>>>(gates, h1h);

    // ---- LSTM layer 1 ----
    k_vec_add<<<4, 256, 0, stream>>>(bsum, bih1, bhh1, 1024);
    k_gemm_wmma<<<dim3(GG / 32, 1024 / 128), 256, 0, stream>>>(
        h1h, wih1h, bsum, gates, GG, 256, 1024, 0);
    k_lstm<<<(GG * HL + 255) / 256, 256, 0, stream>>>(gates, h2h);

    // ---- output projection: out = h2 @ Wp^T + bp  [64 x 512] ----
    k_gemm_wmma<<<dim3(GG / 32, 512 / 128), 256, 0, stream>>>(
        h2h, wph, bp, (float*)d_out, GG, 256, 512, 0);
}
